// EmbedLayer_59304908423194
// MI455X (gfx1250) — matrix operand order repaired and checked by reading, not compiled, not run on hardware
//
#include <hip/hip_runtime.h>
#include <hip/hip_bf16.h>

// out[b,i,h] = sum_j mask[b,i,j] * table[x[b,j] + (i*V+j)*C, h] + bias[i,h]
// Per i: dense GEMM out[:,i,:] = W_i[128x1024] @ T_i[1024x64], where
// W_i[b, j*10 + x[b,j]] = mask[b,i,j] (0/1, exact in f16) and T_i is the
// contiguous table slab rows [i*1000,(i+1)*1000). Padded K rows are harmless:
// W is zero there, and we clamp the table row index so reads stay in bounds.

#define V       100
#define C       10
#define H       64
#define BATCH   128
#define KP      1024     // padded K
#define KT      1000     // true K per slab
#define NSTEP   (KP / 32)
#define BH      64       // batch rows per workgroup (2 halves)
#define BTSTR   40       // padded k-stride of transposed B tile (16B aligned)

typedef __attribute__((ext_vector_type(16))) _Float16 v16h;
typedef __attribute__((ext_vector_type(8)))  _Float16 v8h;
typedef __attribute__((ext_vector_type(2)))  _Float16 v2h;
typedef __attribute__((ext_vector_type(8)))  float    v8f;
typedef __attribute__((ext_vector_type(4)))  float    v4f;

// Stage one 32x64 f32 tile (rows Kbase..Kbase+31, clamped to <KT) into the
// transposed f16 LDS tile Bt[n][k]. Each of 256 threads: 2x global_load_b128
// (coalesced), 4x packed cvt, 4x ds_store_b32.
__device__ __forceinline__
void stage_tile(const float* __restrict__ tblock, _Float16* Btp,
                int Kbase, int tid) {
    const int kp = tid >> 4;          // k-pair index 0..15
    const int n4 = (tid & 15) * 4;    // column group
    int kr0 = Kbase + kp * 2;
    int kr1 = kr0 + 1;
    kr0 = (kr0 < KT) ? kr0 : (KT - 1);   // clamped rows multiply W==0 -> 0
    kr1 = (kr1 < KT) ? kr1 : (KT - 1);
    v4f r0 = *(const v4f*)(tblock + (size_t)kr0 * H + n4);
    v4f r1 = *(const v4f*)(tblock + (size_t)kr1 * H + n4);
    #pragma unroll
    for (int c = 0; c < 4; ++c) {
        v2h p;
        p[0] = (_Float16)r0[c];
        p[1] = (_Float16)r1[c];
        *(v2h*)(Btp + (n4 + c) * BTSTR + kp * 2) = p;
    }
}

__global__ __launch_bounds__(256, 1)
void embed_wmma_kernel(const int* __restrict__ x,
                       const float* __restrict__ mask,
                       const float* __restrict__ table,
                       const float* __restrict__ bias,
                       float* __restrict__ out) {
    __shared__ __attribute__((aligned(16))) _Float16 Wl[BH * KP];        // 128 KB
    __shared__ __attribute__((aligned(16))) _Float16 Btd[2][H * BTSTR];  // 10 KB (double buffer)

    const int i   = blockIdx.x;        // variable index 0..99
    const int bh  = blockIdx.y;        // batch half 0..1
    const int tid = threadIdx.x;

    // ---- Phase 1: zero W (b128 stores), then scatter one-hot mask values ----
    {
        v8h z = {};
        v8h* wv = (v8h*)Wl;
        #pragma unroll
        for (int it = 0; it < (BH * KP) / (8 * 256); ++it)   // 32 iters
            wv[tid + it * 256] = z;
    }
    __syncthreads();
    {
        #pragma unroll
        for (int it = 0; it < (BH * V) / 256; ++it) {        // 25 iters, 6400 scatters
            int idx = tid + it * 256;
            int bl  = idx / V;               // local batch row 0..63
            int j   = idx - bl * V;
            int b   = bh * BH + bl;
            float m = mask[((size_t)b * V + i) * V + j];
            int  xv = x[b * V + j];
            float w = (m == 1.0f) ? 1.0f : 0.0f;
            Wl[bl * KP + j * C + xv] = (_Float16)w;          // ds_store_b16 scatter
        }
    }

    const float* tblock = table + (size_t)i * KT * H;

    // Software pipeline prologue: stage tile 0 (no barrier needed vs Wl writes;
    // the loop's first __syncthreads covers both).
    stage_tile(tblock, Btd[0], 0, tid);

    // ---- Phase 2: K-loop GEMM, double-buffered, one barrier per step ----
    const int wave  = tid >> 5;
    const int lane  = tid & 31;
    const int mt    = wave & 3;          // M tile (16 batch rows)
    const int npair = wave >> 2;         // N tile pair {0,1} or {2,3}
    const int half  = lane >> 4;
    const int l15   = lane & 15;

    v8f acc0 = {};
    v8f acc1 = {};

    for (int kb = 0; kb < NSTEP; ++kb) {
        __syncthreads();                 // buf[kb&1] staged & previous reads done
        const _Float16* Bp = Btd[kb & 1];

        if (kb + 1 < NSTEP)              // stage next tile into other buffer
            stage_tile(tblock, Btd[(kb + 1) & 1], (kb + 1) * 32, tid);
        if (kb + 2 < NSTEP)              // warm L2/WGP$ for the tile after that
            __builtin_prefetch(tblock + (size_t)(kb + 2) * 32 * H + tid * 8, 0, 1);

        const int Kbase = kb * 32;

        // A fragment: 16-bit A 16x32 layout — lane M=l15, K = half*8 + e + (e>=8?8:0)
        const _Float16* wr = Wl + (mt * 16 + l15) * KP + Kbase + half * 8;
        v8h a0 = *(const v8h*)(wr);
        v8h a1 = *(const v8h*)(wr + 16);
        v16h A = __builtin_shufflevector(a0, a1, 0,1,2,3,4,5,6,7,8,9,10,11,12,13,14,15);

        // B fragments: B 32x16 layout — lane N=l15, K = half*8 + e + (e>=8?8:0)
        const _Float16* bp0 = Bp + ((npair * 2 + 0) * 16 + l15) * BTSTR + half * 8;
        const _Float16* bp1 = Bp + ((npair * 2 + 1) * 16 + l15) * BTSTR + half * 8;
        v8h b00 = *(const v8h*)(bp0);
        v8h b01 = *(const v8h*)(bp0 + 16);
        v8h b10 = *(const v8h*)(bp1);
        v8h b11 = *(const v8h*)(bp1 + 16);
        v16h B0 = __builtin_shufflevector(b00, b01, 0,1,2,3,4,5,6,7,8,9,10,11,12,13,14,15);
        v16h B1 = __builtin_shufflevector(b10, b11, 0,1,2,3,4,5,6,7,8,9,10,11,12,13,14,15);

        acc0 = __builtin_amdgcn_wmma_f32_16x16x32_f16(false, A, false, B0,
                                                      (short)0, acc0, false, false);
        acc1 = __builtin_amdgcn_wmma_f32_16x16x32_f16(false, A, false, B1,
                                                      (short)0, acc1, false, false);
    }

    // ---- Phase 3: epilogue — C layout: lane N=l15, VGPR r -> M = r + 8*half ----
    const int n0 = (npair * 2 + 0) * 16 + l15;
    const int n1 = (npair * 2 + 1) * 16 + l15;
    const float bias0 = bias[i * H + n0];
    const float bias1 = bias[i * H + n1];
    #pragma unroll
    for (int r = 0; r < 8; ++r) {
        int M = r + half * 8;
        int b = bh * BH + mt * 16 + M;
        size_t o = ((size_t)b * V + i) * H;
        out[o + n0] = acc0[r] + bias0;
        out[o + n1] = acc1[r] + bias1;
    }
}

extern "C" void kernel_launch(void* const* d_in, const int* in_sizes, int n_in,
                              void* d_out, int out_size, void* d_ws, size_t ws_size,
                              hipStream_t stream) {
    (void)in_sizes; (void)n_in; (void)out_size; (void)d_ws; (void)ws_size;
    const int*   x     = (const int*)d_in[0];
    const float* mask  = (const float*)d_in[1];
    const float* table = (const float*)d_in[2];
    const float* bias  = (const float*)d_in[3];
    float*       out   = (float*)d_out;

    dim3 grid(V, BATCH / BH, 1);   // 100 x 2 workgroups
    embed_wmma_kernel<<<grid, 256, 0, stream>>>(x, mask, table, bias, out);
}
